// HungarianInstanceLoss_64441689309810
// MI455X (gfx1250) — matrix operand order repaired and checked by reading, not compiled, not run on hardware
//
#include <hip/hip_runtime.h>

// ---------------------------------------------------------------------------
// HungarianInstanceLoss for MI455X (gfx1250, wave32, WMMA)
//
// Pass 1 (hil_reduce): one streaming pass over logits[NV][64].
//   Segmented sums Sx[p][j] = sum_{v in inst j} x[p,v] and
//   Sp[p][j] = sum_{v in inst j} sigmoid(x[p,v]) are computed as
//   (64x48) = (64 x K) @ onehot(K x 48) via V_WMMA_F32_16X16X32_F16 with an
//   f16 hi/lo split of A (fp32-exact to ~2^-22), K = 32 voxels per chunk.
//   Per-pred row sums (softplus, sigmoid, focal) ride along in the A-build.
// Pass 2 (hil_finalize): tiny — cost matrices, JV Hungarian, 5 losses.
// ---------------------------------------------------------------------------

typedef __attribute__((ext_vector_type(16))) _Float16 v16h;
typedef __attribute__((ext_vector_type(8)))  float    v8f;

#define HIL_NPRED 64
#define HIL_NINST 48
#define HIL_CHUNK 32
#define HIL_WG    128   // 4 waves; wave w owns preds [16w, 16w+16)

// workspace layout (floats)
#define OFF_SX   0
#define OFF_SP   (HIL_NPRED * HIL_NINST)                 // 3072
#define OFF_RN   (2 * HIL_NPRED * HIL_NINST)             // 6144 softplus rowsum
#define OFF_RP   (OFF_RN + HIL_NPRED)                    // 6208 sigmoid rowsum
#define OFF_RF   (OFF_RP + HIL_NPRED)                    // 6272 focal rowsum
#define OFF_CNT  (OFF_RF + HIL_NPRED)                    // 6336 cnt[2][48]
#define WS_FLOATS (OFF_CNT + 2 * HIL_NINST)              // 6432

__global__ void hil_zero(float* __restrict__ ws, int n) {
  int i = blockIdx.x * blockDim.x + threadIdx.x;
  if (i < n) ws[i] = 0.0f;
}

__global__ __launch_bounds__(HIL_WG)
void hil_reduce(const float* __restrict__ logits,
                const int*   __restrict__ voxel_batch,
                const int*   __restrict__ gt_ids,
                const int*   __restrict__ centroid_batches,
                float*       __restrict__ ws,
                int NV, int nchunks)
{
  __shared__ float    xt[HIL_CHUNK][HIL_NPRED + 1];  // +1: bank-conflict pad
  __shared__ int      gt_s[HIL_CHUNK];
  __shared__ int      vb_s[HIL_CHUNK];
  __shared__ unsigned cnt_s[2 * HIL_NINST];

  const int tid   = threadIdx.x;
  const int wave  = tid >> 5;
  const int lane  = tid & 31;
  const int lhalf = lane >> 4;        // K-half selector in A/C layouts
  const int mloc  = lane & 15;
  const int pred  = wave * 16 + mloc; // this lane's prediction row
  const int cbv   = centroid_batches[pred];

  for (int i = tid; i < 2 * HIL_NINST; i += HIL_WG) cnt_s[i] = 0u;

  // persistent accumulators: 3 N-tiles x (Sx, Sp)
  v8f accX[3], accP[3];
#pragma unroll
  for (int nt = 0; nt < 3; ++nt)
#pragma unroll
    for (int r = 0; r < 8; ++r) { accX[nt][r] = 0.0f; accP[nt][r] = 0.0f; }

  float rn = 0.0f, rp = 0.0f, rf = 0.0f;  // per-lane partial row sums

  for (int chunk = blockIdx.x; chunk < nchunks; chunk += gridDim.x) {
    const long long base = (long long)chunk * HIL_CHUNK;
    __syncthreads();  // protect LDS tile from previous iteration's readers

    // cooperative, coalesced tile load: 32 voxels x 64 preds (float4 units)
    for (int i = tid; i < HIL_CHUNK * 16; i += HIL_WG) {
      int v = i >> 4, c = (i & 15) << 2;
      long long g = base + v;
      float4 val = make_float4(0.f, 0.f, 0.f, 0.f);
      if (g < (long long)NV)
        val = *(const float4*)(logits + g * HIL_NPRED + c);
      xt[v][c + 0] = val.x; xt[v][c + 1] = val.y;
      xt[v][c + 2] = val.z; xt[v][c + 3] = val.w;
    }
    if (tid < HIL_CHUNK) {
      long long g = base + tid;
      int gtv = -1, vbv = -1;
      if (g < (long long)NV) { gtv = gt_ids[g]; vbv = voxel_batch[g]; }
      gt_s[tid] = gtv; vb_s[tid] = vbv;
      if (gtv >= 0 && gtv < HIL_NINST && vbv >= 0 && vbv < 2)
        atomicAdd(&cnt_s[vbv * HIL_NINST + gtv], 1u);
    }
    __syncthreads();

    // Build A fragments (16-bit A-matrix 16x32 layout, ISA 7.12.2):
    // element e -> VGPR r=e>>1, K = (r&3)*2 + (r>>2)*16 + (e&1) + 8*lhalf.
    // While touching each (pred, voxel) element, also accumulate the
    // per-pred transcendental row sums (masked to the pred's batch).
    v16h aXh, aXl, aSh, aSl;
#pragma unroll
    for (int e = 0; e < 16; ++e) {
      int r = e >> 1;
      int K = ((r & 3) << 1) + ((r >> 2) << 4) + (e & 1) + (lhalf << 3);
      float x  = xt[K][pred];
      _Float16 xh = (_Float16)x;
      _Float16 xl = (_Float16)(x - (float)xh);
      float s  = 1.0f / (1.0f + __expf(-x));
      _Float16 sh = (_Float16)s;
      _Float16 sl = (_Float16)(s - (float)sh);
      aXh[e] = xh; aXl[e] = xl; aSh[e] = sh; aSl[e] = sl;
      if (vb_s[K] == cbv) {
        float spv = __logf(1.0f + __expf(-fabsf(x))) + fmaxf(x, 0.0f);
        rn += spv;
        rp += s;
        rf += 0.75f * spv * s * s;
      }
    }

    // B fragment (K x N one-hot): lane%16 = N column, element e = K (voxel, same order as A).
#pragma unroll
    for (int nt = 0; nt < 3; ++nt) {
      v16h bf;
#pragma unroll
      for (int e = 0; e < 16; ++e) {
        int K = (e & 7) + ((e >> 3) << 4) + (lhalf << 3);
        bf[e] = (_Float16)((gt_s[K] == nt * 16 + mloc) ? 1.0f : 0.0f);
      }
      accX[nt] = __builtin_amdgcn_wmma_f32_16x16x32_f16(
          false, aXh, false, bf, (short)0, accX[nt], false, false);
      accX[nt] = __builtin_amdgcn_wmma_f32_16x16x32_f16(
          false, aXl, false, bf, (short)0, accX[nt], false, false);
      accP[nt] = __builtin_amdgcn_wmma_f32_16x16x32_f16(
          false, aSh, false, bf, (short)0, accP[nt], false, false);
      accP[nt] = __builtin_amdgcn_wmma_f32_16x16x32_f16(
          false, aSl, false, bf, (short)0, accP[nt], false, false);
    }
  }

  __syncthreads();
  // Flush. C/D layout: VGPR r, lanes 0-15: M=r; lanes 16-31: M=8+r; N=lane%16.
#pragma unroll
  for (int nt = 0; nt < 3; ++nt) {
#pragma unroll
    for (int r = 0; r < 8; ++r) {
      int M = wave * 16 + r + lhalf * 8;
      int N = nt * 16 + mloc;
      atomicAdd(&ws[OFF_SX + M * HIL_NINST + N], accX[nt][r]);
      atomicAdd(&ws[OFF_SP + M * HIL_NINST + N], accP[nt][r]);
    }
  }
  atomicAdd(&ws[OFF_RN + pred], rn);
  atomicAdd(&ws[OFF_RP + pred], rp);
  atomicAdd(&ws[OFF_RF + pred], rf);
  if (tid < 2 * HIL_NINST)
    atomicAdd(&ws[OFF_CNT + tid], (float)cnt_s[tid]);
}

// ---------------------------------------------------------------------------
// JV Hungarian (e-maxx variant), exact port of the reference, on cost[P][G].
// matchInstOfPred[l] = matched local instance index, or -1.
// ---------------------------------------------------------------------------
__device__ void hil_lsa(const double* cost, int P, int G, int* matchInstOfPred)
{
  const bool transposed = P > G;
  const int n = transposed ? G : P;
  const int m = transposed ? P : G;
  double u[65], v[65], minv[65];
  int    p[65], way[65];
  bool   used[65];
  for (int j = 0; j <= m; ++j) { v[j] = 0.0; p[j] = 0; way[j] = 0; }
  for (int i = 0; i <= n; ++i) u[i] = 0.0;

  for (int i = 1; i <= n; ++i) {
    p[0] = i; int j0 = 0;
    for (int j = 0; j <= m; ++j) { minv[j] = 1e300; used[j] = false; }
    for (;;) {
      used[j0] = true;
      int i0 = p[j0], j1 = 0;
      double delta = 1e300;
      for (int j = 1; j <= m; ++j) {
        if (!used[j]) {
          double cij = transposed ? cost[(j - 1) * G + (i0 - 1)]
                                  : cost[(i0 - 1) * G + (j - 1)];
          double cur = cij - u[i0] - v[j];
          if (cur < minv[j]) { minv[j] = cur; way[j] = j0; }
          if (minv[j] < delta) { delta = minv[j]; j1 = j; }
        }
      }
      for (int j = 0; j <= m; ++j) {
        if (used[j]) { u[p[j]] += delta; v[j] -= delta; }
        else         { minv[j] -= delta; }
      }
      j0 = j1;
      if (p[j0] == 0) break;
    }
    while (j0) { int j1 = way[j0]; p[j0] = p[j1]; j0 = j1; }
  }

  for (int l = 0; l < P; ++l) matchInstOfPred[l] = -1;
  for (int j = 1; j <= m; ++j) {
    if (p[j] > 0) {
      if (transposed) matchInstOfPred[j - 1]    = p[j] - 1; // pred j-1 <-> inst p[j]-1
      else            matchInstOfPred[p[j] - 1] = j - 1;    // pred p[j]-1 <-> inst j-1
    }
  }
}

__global__ __launch_bounds__(32)
void hil_finalize(const float* __restrict__ ws,
                  const int*   __restrict__ centroid_batches,
                  float*       __restrict__ out)
{
  if (threadIdx.x != 0 || blockIdx.x != 0) return;

  const float* Sx   = ws + OFF_SX;
  const float* Sp   = ws + OFF_SP;
  const float* rn   = ws + OFF_RN;
  const float* rpr  = ws + OFF_RP;
  const float* rf   = ws + OFF_RF;
  const float* cntb = ws + OFF_CNT;

  const float L_M = 1.0f, L_U = 7.0f, L_B = 2.0f, L_D = 1.0f;
  float losses[5] = {0.f, 0.f, 0.f, 0.f, 0.f};
  int nbatches = 0;

  double cost[HIL_NPRED * HIL_NINST];
  int pl[HIL_NPRED], gl[HIL_NINST], match[HIL_NPRED];

  for (int b = 0; b < 2; ++b) {
    float nvb = 0.0f;
    for (int j = 0; j < HIL_NINST; ++j) nvb += cntb[b * HIL_NINST + j];
    if (nvb <= 0.0f) continue;   // batch absent from voxel_batch
    nbatches++;

    int P = 0;
    for (int p_ = 0; p_ < HIL_NPRED; ++p_)
      if (centroid_batches[p_] == b) pl[P++] = p_;
    if (P == 0) continue;        // reference: 'continue' (batch still counted)

    int G = 0;
    for (int j = 0; j < HIL_NINST; ++j)
      if (cntb[b * HIL_NINST + j] > 0.5f) gl[G++] = j;

    float bce = 0.f, dice = 0.f, unmatched = 0.f;
    int numUm = P; float fsum = 0.f;
    for (int l = 0; l < P; ++l) match[l] = -1;

    if (G > 0) {
      for (int l = 0; l < P; ++l) {
        int gp = pl[l];
        for (int k = 0; k < G; ++k) {
          int gi = gl[k];
          double cb_ = ((double)rn[gp] - (double)Sx[gp * HIL_NINST + gi]) / (double)nvb;
          double cd_ = 1.0 - (2.0 * (double)Sp[gp * HIL_NINST + gi] + 1.0) /
                             ((double)rpr[gp] + (double)cntb[b * HIL_NINST + gi] + 1.0);
          cost[l * G + k] = (double)L_B * cb_ + (double)L_D * cd_;
        }
      }
      hil_lsa(cost, P, G, match);

      float bsum = 0.f, dsum = 0.f;
      for (int l = 0; l < P; ++l) {
        if (match[l] < 0) continue;
        int gp = pl[l], gi = gl[match[l]];
        bsum += (rn[gp] - Sx[gp * HIL_NINST + gi]) / nvb;
        dsum += 1.0f - (2.0f * Sp[gp * HIL_NINST + gi] + 1.0f) /
                        (rpr[gp] + cntb[b * HIL_NINST + gi] + 1.0f);
      }
      bce  = L_B * bsum / (float)G;
      dice = L_D * dsum / (float)G;
    }

    numUm = 0;
    for (int l = 0; l < P; ++l) {
      if (match[l] < 0) { fsum += rf[pl[l]]; numUm++; }
    }
    if (numUm > 0)
      unmatched = L_U * fsum / ((float)numUm * nvb);

    float matched = L_M * (bce + dice);
    float total = matched + unmatched;
    losses[0] += total; losses[1] += matched; losses[2] += unmatched;
    losses[3] += bce;   losses[4] += dice;
  }

  float inv = (nbatches > 0) ? 1.0f / (float)nbatches : 0.0f;
  for (int i = 0; i < 5; ++i) out[i] = losses[i] * inv;
}

extern "C" void kernel_launch(void* const* d_in, const int* in_sizes, int n_in,
                              void* d_out, int out_size, void* d_ws, size_t ws_size,
                              hipStream_t stream) {
  const float* logits = (const float*)d_in[0];
  const int*   vb     = (const int*)d_in[1];
  const int*   gt     = (const int*)d_in[2];
  const int*   cb     = (const int*)d_in[3];
  float* ws  = (float*)d_ws;
  float* out = (float*)d_out;

  const int NV = in_sizes[1];               // logits is [NV][64]
  const int nchunks = (NV + HIL_CHUNK - 1) / HIL_CHUNK;

  hil_zero<<<(WS_FLOATS + 255) / 256, 256, 0, stream>>>(ws, WS_FLOATS);

  int grids = nchunks < 1024 ? (nchunks > 0 ? nchunks : 1) : 1024;
  hil_reduce<<<grids, HIL_WG, 0, stream>>>(logits, vb, gt, cb, ws, NV, nchunks);

  hil_finalize<<<1, 32, 0, stream>>>(ws, cb, out);
}
